// LipDecoder_53996328845727
// MI455X (gfx1250) — matrix operand order repaired and checked by reading, not compiled, not run on hardware
//
#include <hip/hip_runtime.h>

// ---------------------------------------------------------------------------
// LipDecoder on gfx1250: out = tanh(scale * (X @ W[:, :256]^T + b)),
// scale = 1/(sigma_max(W)+1e-6), plus zeroed (hd, cell) state tail.
// Split-bf16 (hi+lo) v_wmma_f32_16x16x32_bf16 GEMM. M-tile=64/block so each
// B fragment is reused 4x from registers; A fragments come from LDS via
// single ds_load_b128 per fragment-half (padded row stride, bank-clean).
// HBM floor: 64MB x read + 64MB out write ~ 5.5us @ 23.3 TB/s.
// ---------------------------------------------------------------------------

typedef __attribute__((ext_vector_type(16))) __bf16 v16bf;
typedef __attribute__((ext_vector_type(8)))  float  v8f;

union ABfrag { unsigned int u[8]; v16bf v; };

#define A_ROW_STRIDE 260  // dwords: 128 hi pairs + 128 lo pairs + 4 pad (bank skew)
#define A_LDS_DWORDS (64 * A_ROW_STRIDE)

__device__ __forceinline__ unsigned short f32_to_bf16_rn(float f) {
  unsigned int u = __float_as_uint(f);
  u += 0x7FFFu + ((u >> 16) & 1u);
  return (unsigned short)(u >> 16);
}
__device__ __forceinline__ float bf16hi_to_f32(unsigned short h) {
  return __uint_as_float(((unsigned int)h) << 16);
}
// Pack two f32 into a bf16 pair (even K in low 16 bits); residual pair in *lo.
__device__ __forceinline__ unsigned int pack_pair(float f0, float f1, unsigned int* lo) {
  unsigned short h0 = f32_to_bf16_rn(f0);
  unsigned short h1 = f32_to_bf16_rn(f1);
  float r0 = f0 - bf16hi_to_f32(h0);
  float r1 = f1 - bf16hi_to_f32(h1);
  unsigned short l0 = f32_to_bf16_rn(r0);
  unsigned short l1 = f32_to_bf16_rn(r1);
  *lo = (unsigned int)l0 | ((unsigned int)l1 << 16);
  return (unsigned int)h0 | ((unsigned int)h1 << 16);
}
// tanh(z) = 1 - 2/(exp(2z)+1): exact algebraically, saturates to +/-1, NaN-safe.
__device__ __forceinline__ float fast_tanh(float z) {
  float e = __expf(2.0f * z);
  return 1.0f - __fdividef(2.0f, e + 1.0f);
}

// ---------------------------------------------------------------------------
// Kernel 1: G = W * W^T  (256x256), W is 256x512. 32 blocks x 256 threads.
// ---------------------------------------------------------------------------
__global__ __launch_bounds__(256) void lip_gram(const float* __restrict__ W,
                                                float* __restrict__ G) {
  __shared__ float wi[8][512];
  const int i0 = blockIdx.x * 8;
  const int t = threadIdx.x;
  for (int n = 0; n < 16; ++n) {
    int idx = n * 256 + t;  // 0..4095
    wi[idx >> 9][idx & 511] = W[(size_t)i0 * 512 + idx];
  }
  __syncthreads();
  float acc[8] = {};
  const float* wj = W + (size_t)t * 512;
  for (int k = 0; k < 512; ++k) {
    float wv = wj[k];
#pragma unroll
    for (int r = 0; r < 8; ++r) acc[r] += wi[r][k] * wv;
  }
#pragma unroll
  for (int r = 0; r < 8; ++r) G[(size_t)(i0 + r) * 256 + t] = acc[r];
}

// ---------------------------------------------------------------------------
// Kernel 2: power iteration on symmetric G -> scale = 1/(sqrt(lambda)+eps).
// ---------------------------------------------------------------------------
__global__ __launch_bounds__(256) void lip_sigma(const float* __restrict__ G,
                                                 float* __restrict__ scale_out) {
  __shared__ float v[256];
  __shared__ float red[256];
  const int t = threadIdx.x;
  v[t] = 1.0f;
  __syncthreads();
  float nrm = 1.0f;
  for (int it = 0; it < 24; ++it) {
    float u = 0.0f;
    for (int k = 0; k < 256; ++k) u += G[(size_t)k * 256 + t] * v[k];
    red[t] = u * u;
    __syncthreads();
    for (int s = 128; s > 0; s >>= 1) {
      if (t < s) red[t] += red[t + s];
      __syncthreads();
    }
    nrm = sqrtf(red[0]);
    __syncthreads();
    v[t] = u / nrm;
    __syncthreads();
  }
  if (t == 0) {
    float sigma = sqrtf(nrm);  // sigma(W) = sqrt(lambda_max(W W^T))
    scale_out[0] = 1.0f / (sigma + 1e-6f);
  }
}

// ---------------------------------------------------------------------------
// Kernel 3: pre-pack W[:, :256] into per-column-contiguous bf16 hi/lo pairs:
// WpkT[n*128 + q] = bf16 pair (K=2q, K=2q+1) of output column n. A lane's 8
// B-fragment dwords (q = kk*16 + half*8 + j) are contiguous -> 2x b128 loads.
// ---------------------------------------------------------------------------
__global__ __launch_bounds__(128) void lip_packw(const float* __restrict__ W,
                                                 unsigned int* __restrict__ WpkT_hi,
                                                 unsigned int* __restrict__ WpkT_lo) {
  const int n = blockIdx.x;   // 0..255 : output column (row of W)
  const int q = threadIdx.x;  // 0..127 : K-pair index
  float f0 = W[(size_t)n * 512 + 2 * q];
  float f1 = W[(size_t)n * 512 + 2 * q + 1];
  unsigned int lo;
  unsigned int hi = pack_pair(f0, f1, &lo);
  WpkT_hi[n * 128 + q] = hi;
  WpkT_lo[n * 128 + q] = lo;
}

// ---------------------------------------------------------------------------
// Kernel 4: zero the returned (hd, cell) state (8192 floats).
// ---------------------------------------------------------------------------
__global__ void lip_zero_tail(float* __restrict__ out_tail, int n) {
  int i = blockIdx.x * blockDim.x + threadIdx.x;
  if (i < n) out_tail[i] = 0.0f;
}

// ---------------------------------------------------------------------------
// Kernel 5: main GEMM + epilogue. 1024 blocks x 512 threads (16 waves).
// Dynamic LDS A-tile: row-major [row][pair], stride 260 dwords (hi pairs at
// +0..127, lo pairs at +128..255, 4 dwords pad => banks 4r+c, conflict-free).
// Per K-step: 4x global b128 for B (reused over 4 M-subtiles), per subtile
// 4x ds_load_b128 for A, 3 WMMAs (hi*hi, lo*hi, hi*lo).
// ---------------------------------------------------------------------------
__global__ __launch_bounds__(512) void lip_gemm(const float* __restrict__ x,
                                                const float* __restrict__ bias,
                                                const unsigned int* __restrict__ WpkT_hi,
                                                const unsigned int* __restrict__ WpkT_lo,
                                                const float* __restrict__ scale_p,
                                                float* __restrict__ out) {
  extern __shared__ unsigned int Apk[];  // 64 * 260 dwords = 66560 B
  const int tid = threadIdx.x;
  const long mbase = (long)blockIdx.x * 64;

  // Cooperative convert: 64 rows x 256 cols of x -> pair-packed bf16 hi/lo.
  {
    const int row = tid >> 3;          // 0..63
    const int colb = (tid & 7) * 32;   // 32 consecutive floats per thread
    const float4* xp4 = (const float4*)(x + (mbase + row) * 256 + colb);
    unsigned int* rowp = Apk + row * A_ROW_STRIDE;
#pragma unroll
    for (int u = 0; u < 8; ++u) {
      float4 f = xp4[u];
      unsigned int lo0, lo1;
      unsigned int hi0 = pack_pair(f.x, f.y, &lo0);
      unsigned int hi1 = pack_pair(f.z, f.w, &lo1);
      int p = (colb >> 1) + 2 * u;  // K-pair index 0..127
      rowp[p + 0] = hi0;
      rowp[p + 1] = hi1;
      rowp[128 + p + 0] = lo0;
      rowp[128 + p + 1] = lo1;
    }
  }
  __syncthreads();

  const int wave = tid >> 5;   // N-tile index 0..15
  const int lane = tid & 31;
  const int half = lane >> 4;  // lane group
  const int r = lane & 15;
  const int ncol = wave * 16 + r;

  v8f c[4];
#pragma unroll
  for (int m = 0; m < 4; ++m) c[m] = (v8f){0.f, 0.f, 0.f, 0.f, 0.f, 0.f, 0.f, 0.f};

  for (int kk = 0; kk < 8; ++kk) {  // K blocks of 32
    // B (32x16 bf16): lanes 0-15 K 0..15 (pairs per dword), lanes 16-31 K 16..31.
    ABfrag bh, bl;
    {
      const size_t qoff = (size_t)ncol * 128 + kk * 16 + half * 8;
      const uint4* ph = (const uint4*)(WpkT_hi + qoff);
      const uint4* pl = (const uint4*)(WpkT_lo + qoff);
      uint4 h0 = ph[0], h1 = ph[1];
      uint4 l0 = pl[0], l1 = pl[1];
      bh.u[0] = h0.x; bh.u[1] = h0.y; bh.u[2] = h0.z; bh.u[3] = h0.w;
      bh.u[4] = h1.x; bh.u[5] = h1.y; bh.u[6] = h1.z; bh.u[7] = h1.w;
      bl.u[0] = l0.x; bl.u[1] = l0.y; bl.u[2] = l0.z; bl.u[3] = l0.w;
      bl.u[4] = l1.x; bl.u[5] = l1.y; bl.u[6] = l1.z; bl.u[7] = l1.w;
    }
    const int pbase = kk * 16 + half * 8;
#pragma unroll
    for (int m = 0; m < 4; ++m) {
      // A (16x32 bf16): lanes 0-15 K 0..15 +32kk, lanes 16-31 K 16..31 +32kk.
      const unsigned int* ap = Apk + (m * 16 + r) * A_ROW_STRIDE + pbase;
      ABfrag ah, al;
      uint4 a0 = *(const uint4*)(ap + 0);    // hi pairs pbase..pbase+3
      uint4 a1 = *(const uint4*)(ap + 4);    // hi pairs pbase+4..pbase+7
      uint4 a2 = *(const uint4*)(ap + 128);  // lo pairs
      uint4 a3 = *(const uint4*)(ap + 132);
      ah.u[0] = a0.x; ah.u[1] = a0.y; ah.u[2] = a0.z; ah.u[3] = a0.w;
      ah.u[4] = a1.x; ah.u[5] = a1.y; ah.u[6] = a1.z; ah.u[7] = a1.w;
      al.u[0] = a2.x; al.u[1] = a2.y; al.u[2] = a2.z; al.u[3] = a2.w;
      al.u[4] = a3.x; al.u[5] = a3.y; al.u[6] = a3.z; al.u[7] = a3.w;
      c[m] = __builtin_amdgcn_wmma_f32_16x16x32_bf16(false, ah.v, false, bh.v,
                                                     (short)0, c[m], false, false);
      c[m] = __builtin_amdgcn_wmma_f32_16x16x32_bf16(false, al.v, false, bh.v,
                                                     (short)0, c[m], false, false);
      c[m] = __builtin_amdgcn_wmma_f32_16x16x32_bf16(false, ah.v, false, bl.v,
                                                     (short)0, c[m], false, false);
    }
  }

  // Epilogue: C layout VGPR p -> M = m*16 + p + 8*half, N = r.
  const float scale = scale_p[0];
  const float bv = bias[ncol];
#pragma unroll
  for (int m = 0; m < 4; ++m) {
#pragma unroll
    for (int p = 0; p < 8; ++p) {
      long row = mbase + m * 16 + half * 8 + p;
      out[row * 256 + ncol] = fast_tanh(scale * (c[m][p] + bv));
    }
  }
}

// ---------------------------------------------------------------------------
extern "C" void kernel_launch(void* const* d_in, const int* in_sizes, int n_in,
                              void* d_out, int out_size, void* d_ws, size_t ws_size,
                              hipStream_t stream) {
  const float* x = (const float*)d_in[0];  // [16,4096,256]
  const float* W = (const float*)d_in[1];  // [256,512]
  const float* b = (const float*)d_in[2];  // [256]
  float* out = (float*)d_out;              // [16,4096,256] + 2*[1,16,256]

  char* ws = (char*)d_ws;
  float* scale_p = (float*)ws;                                   // 4 B
  unsigned int* WpkT_hi = (unsigned int*)(ws + 1024);            // 128 KB
  unsigned int* WpkT_lo = (unsigned int*)(ws + 1024 + 131072);   // 128 KB
  float* G = (float*)(ws + 1024 + 2 * 131072);                   // 256 KB

  lip_gram<<<32, 256, 0, stream>>>(W, G);
  lip_sigma<<<1, 256, 0, stream>>>(G, scale_p);
  lip_packw<<<256, 128, 0, stream>>>(W, WpkT_hi, WpkT_lo);
  lip_zero_tail<<<32, 256, 0, stream>>>(out + (size_t)65536 * 256, 8192);
  lip_gemm<<<1024, 512, A_LDS_DWORDS * 4, stream>>>(x, b, WpkT_hi, WpkT_lo,
                                                    scale_p, out);
}
